// DeltaNet_39359080300553
// MI455X (gfx1250) — matrix operand order repaired and checked by reading, not compiled, not run on hardware
//
#include <hip/hip_runtime.h>
#include <hip/hip_bf16.h>
#include <math.h>

#define HID 1024
#define NH 4
#define HD 256
#define KS 4
#define CCH 64            // chunk size C
#define BB 4
#define LL 4096
#define NCH (LL / CCH)    // 64 chunks
#define ML (BB * LL)      // 16384 rows
#define EPSV 1e-5f

typedef __attribute__((ext_vector_type(16))) _Float16 v16h;
typedef __attribute__((ext_vector_type(8)))  float    v8f;

__device__ __forceinline__ float sigmoidf_(float x) { return 1.0f / (1.0f + __expf(-x)); }

__device__ __forceinline__ v8f wmma_f16(v16h a, v16h b, v8f c) {
  return __builtin_amdgcn_wmma_f32_16x16x32_f16(false, a, false, b, (short)0, c, false, false);
}

// A-fragment k index (contiguous: k = e + 16*hi, matching B)
__device__ __forceinline__ int a_kidx(int e, int hi) {
  return (hi << 4) + e;
}

// ---------------- f32 -> f16 convert ----------------
__global__ void cvt_f32_to_f16(const float* __restrict__ src, _Float16* __restrict__ dst, int n) {
  int i = blockIdx.x * blockDim.x + threadIdx.x;
  if (i < n) dst[i] = (_Float16)src[i];
}

// ---------------- generic WMMA GEMM: C[M,N] = A[M,K] * B[K,N], f16 in, OUT_T out ------
// Each wave computes a 16x64 tile (1x4 WMMA tiles): A fragment reused 4x.
// A-frag: lane m = lane&15, k per a_kidx. B-frag: lane n = lane&15, k = e + 16*(lane>=16).
// D: lane n = lane&15, vgpr r: m = r + 8*(lane>=16).
template <typename OUT_T>
__global__ __launch_bounds__(256) void gemm_nn(const _Float16* __restrict__ A,
                                               const _Float16* __restrict__ B,
                                               OUT_T* __restrict__ C,
                                               int M, int N, int K) {
  const int lane = threadIdx.x & 31;
  const int wave = threadIdx.x >> 5;
  const int groupsN = N >> 6;                    // 64-wide column groups
  int wg = blockIdx.x * 8 + wave;
  int tm = wg / groupsN;
  int tg = wg - tm * groupsN;
  if (tm >= (M >> 4)) return;
  const int hi = lane >> 4;
  const int rowA = tm * 16 + (lane & 15);
  const int col0 = tg * 64 + (lane & 15);
  v8f acc0 = {}, acc1 = {}, acc2 = {}, acc3 = {};
  for (int kk = 0; kk < K; kk += 32) {
    v16h a, b0, b1, b2, b3;
#pragma unroll
    for (int e = 0; e < 16; ++e) {
      a[e] = A[(size_t)rowA * K + kk + a_kidx(e, hi)];
      const _Float16* brow = B + (size_t)(kk + e + (hi << 4)) * N + col0;
      b0[e] = brow[0];
      b1[e] = brow[16];
      b2[e] = brow[32];
      b3[e] = brow[48];
    }
    acc0 = wmma_f16(a, b0, acc0);
    acc1 = wmma_f16(a, b1, acc1);
    acc2 = wmma_f16(a, b2, acc2);
    acc3 = wmma_f16(a, b3, acc3);
  }
#pragma unroll
  for (int r = 0; r < 8; ++r) {
    int m = tm * 16 + r + (hi << 3);
    OUT_T* crow = C + (size_t)m * N + col0;
    crow[0]  = (OUT_T)acc0[r];
    crow[16] = (OUT_T)acc1[r];
    crow[32] = (OUT_T)acc2[r];
    crow[48] = (OUT_T)acc3[r];
  }
}

// ---------------- per-chunk A = Kb(64x1024) * K^T; epilogue T0 = I - tril(A,-1) -------
// One wave per (batch, tm): computes full 16x64 row band. NT form: B[k][j] = K[j][k].
__global__ __launch_bounds__(256) void a_chunk_gemm(const _Float16* __restrict__ Kbh,
                                                    const _Float16* __restrict__ Kh,
                                                    float* __restrict__ Tbuf) {
  const int lane = threadIdx.x & 31;
  const int wave = threadIdx.x >> 5;
  int wg = blockIdx.x * 8 + wave;       // 1024 waves: 256 batches * 4 row-bands
  int batch = wg >> 2;                  // b*NCH + n
  int tm = wg & 3;
  const _Float16* Ab = Kbh + (size_t)batch * CCH * HID;
  const _Float16* Bt = Kh + (size_t)batch * CCH * HID;
  const int hi = lane >> 4;
  const int rowA = tm * 16 + (lane & 15);
  const int colL = lane & 15;
  v8f acc0 = {}, acc1 = {}, acc2 = {}, acc3 = {};
  for (int kk = 0; kk < HID; kk += 32) {
    v16h a, b0, b1, b2, b3;
#pragma unroll
    for (int e = 0; e < 16; ++e) {
      a[e] = Ab[(size_t)rowA * HID + kk + a_kidx(e, hi)];
      int kb = kk + e + (hi << 4);
      b0[e] = Bt[(size_t)(colL)      * HID + kb];
      b1[e] = Bt[(size_t)(colL + 16) * HID + kb];
      b2[e] = Bt[(size_t)(colL + 32) * HID + kb];
      b3[e] = Bt[(size_t)(colL + 48) * HID + kb];
    }
    acc0 = wmma_f16(a, b0, acc0);
    acc1 = wmma_f16(a, b1, acc1);
    acc2 = wmma_f16(a, b2, acc2);
    acc3 = wmma_f16(a, b3, acc3);
  }
  float* Tb = Tbuf + (size_t)batch * CCH * CCH;
#pragma unroll
  for (int r = 0; r < 8; ++r) {
    int gm = tm * 16 + r + (hi << 3);
    float* trow = Tb + gm * CCH;
    int g0 = colL, g1 = colL + 16, g2 = colL + 32, g3 = colL + 48;
    trow[g0] = ((gm == g0) ? 1.0f : 0.0f) - ((g0 < gm) ? acc0[r] : 0.0f);
    trow[g1] = ((gm == g1) ? 1.0f : 0.0f) - ((g1 < gm) ? acc1[r] : 0.0f);
    trow[g2] = ((gm == g2) ? 1.0f : 0.0f) - ((g2 < gm) ? acc2[r] : 0.0f);
    trow[g3] = ((gm == g3) ? 1.0f : 0.0f) - ((g3 < gm) ? acc3[r] : 0.0f);
  }
}

// ---------------- per-chunk W/U = T(64x64) * M(64x1024), 1x4 wave tiles ----------------
__global__ __launch_bounds__(256) void wu_gemm(const _Float16* __restrict__ Th,
                                               const _Float16* __restrict__ Bm,
                                               float* __restrict__ Out) {
  const int lane = threadIdx.x & 31;
  const int wave = threadIdx.x >> 5;
  int wg = blockIdx.x * 8 + wave;       // 16384 waves: 256 batches * 4 tm * 16 groups
  int batch = wg >> 6;
  int tile = wg & 63;
  int tm = tile >> 4, tg = tile & 15;
  const _Float16* Ab = Th + (size_t)batch * CCH * CCH;
  const _Float16* Bb_ = Bm + (size_t)batch * CCH * HID;
  float* Ob = Out + (size_t)batch * CCH * HID;
  const int hi = lane >> 4;
  const int rowA = tm * 16 + (lane & 15);
  const int col0 = tg * 64 + (lane & 15);
  v8f acc0 = {}, acc1 = {}, acc2 = {}, acc3 = {};
  for (int kk = 0; kk < CCH; kk += 32) {
    v16h a, b0, b1, b2, b3;
#pragma unroll
    for (int e = 0; e < 16; ++e) {
      a[e] = Ab[rowA * CCH + kk + a_kidx(e, hi)];
      const _Float16* brow = Bb_ + (size_t)(kk + e + (hi << 4)) * HID + col0;
      b0[e] = brow[0];
      b1[e] = brow[16];
      b2[e] = brow[32];
      b3[e] = brow[48];
    }
    acc0 = wmma_f16(a, b0, acc0);
    acc1 = wmma_f16(a, b1, acc1);
    acc2 = wmma_f16(a, b2, acc2);
    acc3 = wmma_f16(a, b3, acc3);
  }
#pragma unroll
  for (int r = 0; r < 8; ++r) {
    int m = tm * 16 + r + (hi << 3);
    float* crow = Ob + (size_t)m * HID + col0;
    crow[0]  = acc0[r];
    crow[16] = acc1[r];
    crow[32] = acc2[r];
    crow[48] = acc3[r];
  }
}

// ---------------- beta = sigmoid(x @ Wb) ----------------
__global__ void beta_kernel(const float* __restrict__ x, const float* __restrict__ Wb,
                            float* __restrict__ beta) {
  int idx = blockIdx.x * blockDim.x + threadIdx.x;   // ML*NH
  if (idx >= ML * NH) return;
  int row = idx >> 2, hh = idx & 3;
  const float* xr = x + (size_t)row * HID;
  float acc = 0.f;
  for (int k = 0; k < HID; ++k) acc += xr[k] * Wb[k * NH + hh];
  beta[idx] = sigmoidf_(acc);
}

// ---------------- causal depthwise conv (K=4, left pad) + SiLU ----------------
__device__ __forceinline__ float dwconv_silu(const _Float16* pre, const float* w,
                                             int b, int l, int ch) {
  float acc = 0.f;
#pragma unroll
  for (int t = 0; t < KS; ++t) {
    int ls = l - (KS - 1) + t;
    if (ls >= 0) acc += w[ch * KS + t] * (float)pre[(size_t)(b * LL + ls) * HID + ch];
  }
  return acc * sigmoidf_(acc);
}

__global__ __launch_bounds__(256) void conv_q_kernel(const _Float16* __restrict__ pre,
                                                     const float* __restrict__ w,
                                                     float* __restrict__ qn) {
  __shared__ float red[256];
  int bid = blockIdx.x;                 // ((b*LL + l)*NH + h)
  int h = bid & 3;
  int l = (bid >> 2) & (LL - 1);
  int b = bid >> 14;
  int d = threadIdx.x;
  int ch = h * HD + d;
  float y = dwconv_silu(pre, w, b, l, ch);
  red[d] = y * y;
  __syncthreads();
  for (int s = 128; s > 0; s >>= 1) { if (d < s) red[d] += red[d + s]; __syncthreads(); }
  float inv = rsqrtf(red[0]);
  qn[(size_t)(b * LL + l) * HID + ch] = y * inv;
}

__global__ __launch_bounds__(256) void conv_k_kernel(const _Float16* __restrict__ pre,
                                                     const float* __restrict__ w,
                                                     const float* __restrict__ beta,
                                                     float* __restrict__ kn,
                                                     _Float16* __restrict__ kh,
                                                     _Float16* __restrict__ kbh) {
  __shared__ float red[256];
  int bid = blockIdx.x;
  int h = bid & 3;
  int l = (bid >> 2) & (LL - 1);
  int b = bid >> 14;
  int d = threadIdx.x;
  int ch = h * HD + d;
  float y = dwconv_silu(pre, w, b, l, ch);
  red[d] = y * y;
  __syncthreads();
  for (int s = 128; s > 0; s >>= 1) { if (d < s) red[d] += red[d + s]; __syncthreads(); }
  float val = y * rsqrtf(red[0]);
  size_t o = (size_t)(b * LL + l) * HID + ch;
  kn[o] = val;
  kh[o] = (_Float16)val;
  float bt = beta[(size_t)(b * LL + l) * NH + h];
  kbh[o] = (_Float16)(val * bt);
}

__global__ __launch_bounds__(256) void conv_v_kernel(const _Float16* __restrict__ pre,
                                                     const float* __restrict__ w,
                                                     const float* __restrict__ beta,
                                                     _Float16* __restrict__ vbh) {
  int bid = blockIdx.x;
  int h = bid & 3;
  int l = (bid >> 2) & (LL - 1);
  int b = bid >> 14;
  int d = threadIdx.x;
  int ch = h * HD + d;
  float y = dwconv_silu(pre, w, b, l, ch);
  float bt = beta[(size_t)(b * LL + l) * NH + h];
  vbh[(size_t)(b * LL + l) * HID + ch] = (_Float16)(y * bt);
}

// ---------------- masked rowdot: rAi[b,l] = (l%C <= b) ? sum(q*k) : 0 ----------------
__global__ __launch_bounds__(256) void rowdot_kernel(const float* __restrict__ qn,
                                                     const float* __restrict__ kn,
                                                     float* __restrict__ rAi) {
  __shared__ float red[256];
  int row = blockIdx.x;
  int t = threadIdx.x;
  const float* qr = qn + (size_t)row * HID;
  const float* kr = kn + (size_t)row * HID;
  float acc = 0.f;
  for (int i = t; i < HID; i += 256) acc += qr[i] * kr[i];
  red[t] = acc; __syncthreads();
  for (int s = 128; s > 0; s >>= 1) { if (t < s) red[t] += red[t + s]; __syncthreads(); }
  if (t == 0) {
    int b = row / LL;
    int c = row & (CCH - 1);
    rAi[row] = (c <= b) ? red[0] : 0.0f;
  }
}

// ---------------- T-inverse recurrence (literal reference body) ----------------
__global__ __launch_bounds__(64) void tinv_kernel(const float* __restrict__ Tbuf,
                                                  _Float16* __restrict__ Th) {
  __shared__ float T[CCH * CCH];
  int batch = blockIdx.x;
  int c = threadIdx.x;
  for (int i = c; i < CCH * CCH; i += 64) T[i] = Tbuf[(size_t)batch * CCH * CCH + i];
  __syncthreads();
  for (int i = 1; i < CCH; ++i) {
    float acc = 0.f;
    for (int j = 0; j < CCH; ++j) acc += T[i * CCH + j] * T[j * CCH + c];
    __syncthreads();
    if (c < i) T[i * CCH + c] += acc;
    __syncthreads();
  }
  for (int i = c; i < CCH * CCH; i += 64) Th[(size_t)batch * CCH * CCH + i] = (_Float16)T[i];
}

// ---------------- sequential chunk scan: S in regs (64/thread), u tile in LDS --------
__global__ __launch_bounds__(1024) void scan_kernel(const float* __restrict__ qn,
                                                    const float* __restrict__ kn,
                                                    const float* __restrict__ Wv,
                                                    const float* __restrict__ Uv,
                                                    const float* __restrict__ rAi,
                                                    float* __restrict__ o,
                                                    float* __restrict__ Sout) {
  __shared__ float u_lds[CCH * HD];   // 64 KB
  __shared__ float sd_lds[HD];
  const int bh = blockIdx.x;          // b*NH + h
  const int b = bh >> 2, h = bh & 3;
  const int tid = threadIdx.x;
  const int d = tid >> 2;             // state row
  const int eq = tid & 3;
  const int e0 = eq * 64;             // state col range [e0, e0+64)
  float s[64];
#pragma unroll
  for (int j = 0; j < 64; ++j) s[j] = 0.f;
  for (int n = 0; n < NCH; ++n) {
    if ((d >> 6) == eq) sd_lds[d] = s[d & 63];   // diag(S)
    __syncthreads();
#pragma unroll 4
    for (int t = 0; t < 16; ++t) {
      int idx = tid + t * 1024;
      int c = idx >> 8;
      int ch = idx & 255;
      int gl = n * CCH + c;
      size_t base = ((size_t)(b * LL + gl) * NH + h) * HD + ch;
      float sd = sd_lds[ch];
      float uval = Uv[base] - Wv[base] * sd;
      float ai = rAi[(size_t)b * LL + gl];
      o[base] = qn[base] * sd + ai * uval;
      u_lds[c * HD + ch] = uval;
    }
    __syncthreads();
    for (int c = 0; c < CCH; ++c) {
      float kc = kn[((size_t)(b * LL + n * CCH + c) * NH + h) * HD + d];
      const float4* up = (const float4*)&u_lds[c * HD + e0];
#pragma unroll
      for (int j4 = 0; j4 < 16; ++j4) {
        float4 uv = up[j4];
        s[4 * j4 + 0] += kc * uv.x;
        s[4 * j4 + 1] += kc * uv.y;
        s[4 * j4 + 2] += kc * uv.z;
        s[4 * j4 + 3] += kc * uv.w;
      }
    }
    __syncthreads();
  }
#pragma unroll
  for (int j = 0; j < 64; ++j)
    Sout[((size_t)(b * NH + h) * HD + d) * HD + e0 + j] = s[j];
}

// ---------------- RMSNorm over head_dim with g, output f16 ----------------
__global__ __launch_bounds__(256) void rmsnorm_kernel(const float* __restrict__ o,
                                                      const float* __restrict__ g,
                                                      _Float16* __restrict__ oh) {
  __shared__ float red[256];
  int bid = blockIdx.x;               // (b*LL + l)*NH + h
  int h = bid & 3;
  int rl = bid >> 2;                  // b*LL + l
  int dd = threadIdx.x;
  size_t base = (size_t)rl * HID + h * HD + dd;
  float v = o[base];
  red[dd] = v * v;
  __syncthreads();
  for (int s = 128; s > 0; s >>= 1) { if (dd < s) red[dd] += red[dd + s]; __syncthreads(); }
  float inv = rsqrtf(red[0] / (float)HD + EPSV);
  oh[base] = (_Float16)(v * inv * g[dd]);
}

extern "C" void kernel_launch(void* const* d_in, const int* in_sizes, int n_in,
                              void* d_out, int out_size, void* d_ws, size_t ws_size,
                              hipStream_t stream) {
  const float* x  = (const float*)d_in[0];
  const float* Wq = (const float*)d_in[1];
  const float* Wk = (const float*)d_in[2];
  const float* Wv = (const float*)d_in[3];
  const float* Wb = (const float*)d_in[4];
  const float* cq = (const float*)d_in[5];
  const float* ck = (const float*)d_in[6];
  const float* cv = (const float*)d_in[7];
  const float* g  = (const float*)d_in[8];
  const float* Wo = (const float*)d_in[9];

  char* ws = (char*)d_ws;
  size_t off = 0;
  auto alloc = [&](size_t bytes) -> void* {
    void* p = ws + off;
    off += (bytes + 255) & ~(size_t)255;
    return p;
  };
  _Float16* XH   = (_Float16*)alloc((size_t)ML * HID * 2);
  _Float16* WQH  = (_Float16*)alloc((size_t)HID * HID * 2);
  _Float16* WKH  = (_Float16*)alloc((size_t)HID * HID * 2);
  _Float16* WVH  = (_Float16*)alloc((size_t)HID * HID * 2);
  _Float16* WOH  = (_Float16*)alloc((size_t)HID * HID * 2);
  _Float16* QPRE = (_Float16*)alloc((size_t)ML * HID * 2);
  _Float16* KPRE = (_Float16*)alloc((size_t)ML * HID * 2);
  _Float16* VPRE = (_Float16*)alloc((size_t)ML * HID * 2);
  float*    BETA = (float*)   alloc((size_t)ML * NH * 4);
  float*    QN   = (float*)   alloc((size_t)ML * HID * 4);
  float*    KN   = (float*)   alloc((size_t)ML * HID * 4);
  _Float16* KH   = (_Float16*)alloc((size_t)ML * HID * 2);
  _Float16* KBH  = (_Float16*)alloc((size_t)ML * HID * 2);
  _Float16* VBH  = (_Float16*)alloc((size_t)ML * HID * 2);
  float*    RAI  = (float*)   alloc((size_t)ML * 4);
  float*    TB   = (float*)   alloc((size_t)BB * NCH * CCH * CCH * 4);
  _Float16* TH   = (_Float16*)alloc((size_t)BB * NCH * CCH * CCH * 2);
  float*    WW   = (float*)   alloc((size_t)ML * HID * 4);
  float*    UU   = (float*)   alloc((size_t)ML * HID * 4);
  float*    OO   = (float*)   alloc((size_t)ML * HID * 4);
  _Float16* OH   = (_Float16*)alloc((size_t)ML * HID * 2);

  float* out_main = (float*)d_out;                 // (B, L, HID)
  float* out_S    = out_main + (size_t)ML * HID;   // (B, NH, HD, HD)

  // 1. conversions to f16
  cvt_f32_to_f16<<<(ML * HID + 255) / 256, 256, 0, stream>>>(x, XH, ML * HID);
  cvt_f32_to_f16<<<(HID * HID + 255) / 256, 256, 0, stream>>>(Wq, WQH, HID * HID);
  cvt_f32_to_f16<<<(HID * HID + 255) / 256, 256, 0, stream>>>(Wk, WKH, HID * HID);
  cvt_f32_to_f16<<<(HID * HID + 255) / 256, 256, 0, stream>>>(Wv, WVH, HID * HID);
  cvt_f32_to_f16<<<(HID * HID + 255) / 256, 256, 0, stream>>>(Wo, WOH, HID * HID);

  // 2. projections (WMMA, 16x64 per wave): (16384 x 1024) @ (1024 x 1024)
  const int gemm_grid = (ML / 16) * (HID / 64) / 8;   // 2048 blocks
  gemm_nn<_Float16><<<gemm_grid, 256, 0, stream>>>(XH, WQH, QPRE, ML, HID, HID);
  gemm_nn<_Float16><<<gemm_grid, 256, 0, stream>>>(XH, WKH, KPRE, ML, HID, HID);
  gemm_nn<_Float16><<<gemm_grid, 256, 0, stream>>>(XH, WVH, VPRE, ML, HID, HID);

  // 3. beta
  beta_kernel<<<(ML * NH + 255) / 256, 256, 0, stream>>>(x, Wb, BETA);

  // 4. conv + SiLU (+norm, +beta products)
  conv_q_kernel<<<ML * NH, 256, 0, stream>>>(QPRE, cq, QN);
  conv_k_kernel<<<ML * NH, 256, 0, stream>>>(KPRE, ck, BETA, KN, KH, KBH);
  conv_v_kernel<<<ML * NH, 256, 0, stream>>>(VPRE, cv, BETA, VBH);

  // 5. masked q.k rowdot (reference tril-(B,C) quirk)
  rowdot_kernel<<<ML, 256, 0, stream>>>(QN, KN, RAI);

  // 6. per-chunk A = Kb K^T -> T0 = I - tril(A,-1)   (WMMA NT, 16x64 per wave)
  a_chunk_gemm<<<(BB * NCH * 4) / 8, 256, 0, stream>>>(KBH, KH, TB);

  // 7. T recurrence (forward substitution), emit f16 T
  tinv_kernel<<<BB * NCH, 64, 0, stream>>>(TB, TH);

  // 8. W = T Kb, U = T Vb   (WMMA, batched, 16x64 per wave)
  wu_gemm<<<(BB * NCH * 64) / 8, 256, 0, stream>>>(TH, KBH, WW);
  wu_gemm<<<(BB * NCH * 64) / 8, 256, 0, stream>>>(TH, VBH, UU);

  // 9. sequential chunk scan -> o and final S (tuple tail of d_out)
  scan_kernel<<<BB * NH, 1024, 0, stream>>>(QN, KN, WW, UU, RAI, OO, out_S);

  // 10. RMSNorm * g -> f16
  rmsnorm_kernel<<<ML * NH, 256, 0, stream>>>(OO, g, OH);

  // 11. out = o @ Wo  (WMMA, f32 out directly to d_out)
  gemm_nn<float><<<gemm_grid, 256, 0, stream>>>(OH, WOH, out_main, ML, HID, HID);
}
